// RingAttention_13563506721408
// MI455X (gfx1250) — matrix operand order repaired and checked by reading, not compiled, not run on hardware
//
#include <hip/hip_runtime.h>

// ---- CDNA5 (gfx1250) vector types for WMMA -------------------------------
typedef __bf16 v16bf __attribute__((ext_vector_type(16)));
typedef __bf16 v8bf  __attribute__((ext_vector_type(8)));
typedef __bf16 v4bf  __attribute__((ext_vector_type(4)));
typedef float  v8f   __attribute__((ext_vector_type(8)));
typedef float  v4f   __attribute__((ext_vector_type(4)));

// ---- problem constants (match reference setup_inputs) --------------------
constexpr int B  = 2;
constexpr int S  = 2048;
constexpr int H  = 16;
constexpr int D  = 128;
constexpr int W  = 4;
constexpr int KV = W * S;            // 8192 total keys per (b,h)
constexpr int MT = 128;              // query rows per workgroup (8 waves x 16)
constexpr int NT = 64;               // keys per KV tile
constexpr int NWAVE = MT / 16;       // 8 waves of 32
constexpr int LKD = D + 8;           // padded stride, bf16 K rows
constexpr int LVN = NT + 8;          // padded stride, bf16 V^T rows
constexpr int LPN = NT + 8;          // padded stride, P rows
constexpr float SCALE = 0.08838834764831845f;   // D^-0.5

// ---- dynamic-LDS layout (bytes) ------------------------------------------
// fp32 async staging tiles (landing zone for tile t+1), then bf16 compute bufs
constexpr unsigned OFF_KF = 0;                               // 64*128*4 = 32768
constexpr unsigned OFF_VF = OFF_KF + NT * D * 4;             // 32768
constexpr unsigned OFF_KB = OFF_VF + NT * D * 4;             // 65536 (bf16 K, 17408)
constexpr unsigned OFF_VB = OFF_KB + NT * LKD * 2;           // 82944 (bf16 V^T, 18432)
constexpr unsigned OFF_P  = OFF_VB + D * LVN * 2;            // 101376 (P, 18432)
constexpr unsigned SMEM_BYTES = OFF_P + NWAVE * 16 * LPN * 2; // 119808

// ---- CDNA5 async copy: global -> LDS, tracked by ASYNCcnt ----------------
__device__ __forceinline__ void async_load_b128(unsigned lds_off,
                                                const float* sbase,
                                                unsigned goff_bytes) {
    asm volatile("global_load_async_to_lds_b128 %0, %1, %2 offset:0"
                 :
                 : "v"(lds_off), "v"(goff_bytes), "s"(sbase)
                 : "memory");
}
__device__ __forceinline__ void wait_asynccnt0() {
    asm volatile("s_wait_asynccnt 0" ::: "memory");
}

__global__ __launch_bounds__(256)
void ring_attn_fwd(const float* __restrict__ qg, const float* __restrict__ kg,
                   const float* __restrict__ vg, float* __restrict__ og)
{
    extern __shared__ char smem[];
    float* kf = (float*)(smem + OFF_KF);                    // stage K fp32 [64][128]
    float* vf = (float*)(smem + OFF_VF);                    // stage V fp32 [64][128]
    __bf16 (*lds_k)[LKD] = (__bf16(*)[LKD])(smem + OFF_KB); // K tile [key][d]
    __bf16 (*lds_v)[LVN] = (__bf16(*)[LVN])(smem + OFF_VB); // V^T   [d][key]
    __bf16 (*lds_p)[16][LPN] = (__bf16(*)[16][LPN])(smem + OFF_P);
    // flat LDS addresses carry the byte offset in bits [31:0]
    const unsigned ldsbase = (unsigned)(size_t)smem;

    const int tid  = threadIdx.x;
    const int wave = tid >> 5;
    const int lane = tid & 31;
    const int half = lane >> 4;     // wave32 half (WMMA lane halves)
    const int l16  = lane & 15;

    const int qtiles = S / MT;                       // 16
    const int qt = blockIdx.x % qtiles;
    const int h  = (blockIdx.x / qtiles) % H;
    const int b  = blockIdx.x / (qtiles * H);
    const int qbase = qt * MT;

    const float* kb0 = kg + ((size_t)b * KV * H + h) * D;
    const float* vb0 = vg + ((size_t)b * KV * H + h) * D;
    const unsigned rowbytes = (unsigned)(H * D * 4);   // 8192 B between keys

    // ---- Q fragments in WMMA A layout (16x32 bf16 per chunk), scale folded
    // A layout (16-bit, 16x32): lane(0-15)=row M; half0 holds K 0-7 & 16-23,
    // half1 holds K 8-15 & 24-31 (elements [0..7],[8..15] of the v16bf).
    v16bf aq[4];
    {
        const int qrow = qbase + wave * 16 + l16;
        const float* qp = qg + ((size_t)(b * S + qrow) * H + h) * D;
#pragma unroll
        for (int c = 0; c < 4; ++c) {
            const int d0 = 32 * c + 8 * half;
            const int d1 = d0 + 16;
            v4f f0 = *(const v4f*)(qp + d0);
            v4f f1 = *(const v4f*)(qp + d0 + 4);
            v4f f2 = *(const v4f*)(qp + d1);
            v4f f3 = *(const v4f*)(qp + d1 + 4);
#pragma unroll
            for (int j = 0; j < 4; ++j) {
                aq[c][j]      = (__bf16)(f0[j] * SCALE);
                aq[c][4 + j]  = (__bf16)(f1[j] * SCALE);
                aq[c][8 + j]  = (__bf16)(f2[j] * SCALE);
                aq[c][12 + j] = (__bf16)(f3[j] * SCALE);
            }
        }
    }

    // ---- accumulators: 16x128 output tile per wave + online-softmax state
    v8f   o_acc[8];
    float m_run[8], l_run[8];
#pragma unroll
    for (int dg = 0; dg < 8; ++dg)
#pragma unroll
        for (int j = 0; j < 8; ++j) o_acc[dg][j] = 0.f;
#pragma unroll
    for (int i = 0; i < 8; ++i) { m_run[i] = -INFINITY; l_run[i] = 0.f; }

    // ---- async stage of one fp32 KV tile (16 x b128 per thread, no VGPR data)
    auto stage_async = [&](int kb) {
#pragma unroll
        for (int i = 0; i < (NT * D * 4 / 16) / 256; ++i) {   // 8 chunks/thread
            const int idx = tid + 256 * i;
            const int key = idx >> 5;            // 0..63
            const int ch  = idx & 31;            // 16B chunk within row
            const unsigned go = (unsigned)(kb + key) * rowbytes + ch * 16u;
            async_load_b128(ldsbase + OFF_KF + key * (D * 4) + ch * 16, kb0, go);
            async_load_b128(ldsbase + OFF_VF + key * (D * 4) + ch * 16, vb0, go);
        }
    };
    // ---- LDS->LDS convert: fp32 stage -> bf16 K (row-major) + V (transposed)
    auto convert_stage = [&]() {
#pragma unroll
        for (int i = 0; i < (NT * D / 4) / 256; ++i) {        // 8 vec4/thread
            const int idx = tid + 256 * i;
            const int key = idx >> 5;
            const int dq  = (idx & 31) * 4;
            v4f fkv = *(const v4f*)(kf + key * D + dq);
            v4f fvv = *(const v4f*)(vf + key * D + dq);
            v4bf bk;
#pragma unroll
            for (int j = 0; j < 4; ++j) bk[j] = (__bf16)fkv[j];
            *(v4bf*)&lds_k[key][dq] = bk;
#pragma unroll
            for (int j = 0; j < 4; ++j) lds_v[dq + j][key] = (__bf16)fvv[j];
        }
    };

    // prologue: fill tile 0
    stage_async(0);
    wait_asynccnt0();
    __syncthreads();
    convert_stage();
    __syncthreads();

    constexpr int NTILES = KV / NT;                 // 128
    for (int t = 0; t < NTILES; ++t) {
        // kick off tile t+1 into the fp32 stage while we compute tile t
        if (t + 1 < NTILES) stage_async((t + 1) * NT);

        // ---- S = (Q*scale) K^T : 4 col groups x 4 K-chunks of WMMA
        // B layout (16-bit, 32x16): lane=column N; half0 K=0..7 & 16..23, half1 K=8..15 & 24..31
        v8f sc[4];
#pragma unroll
        for (int g = 0; g < 4; ++g) {
            v8f s = {0.f, 0.f, 0.f, 0.f, 0.f, 0.f, 0.f, 0.f};
#pragma unroll
            for (int c = 0; c < 4; ++c) {
                const __bf16* kp = &lds_k[g * 16 + l16][32 * c + 8 * half];
                v8bf b0 = *(const v8bf*)kp;
                v8bf b1 = *(const v8bf*)(kp + 16);
                v16bf bb;
#pragma unroll
                for (int j = 0; j < 8; ++j) { bb[j] = b0[j]; bb[8 + j] = b1[j]; }
                s = __builtin_amdgcn_wmma_f32_16x16x32_bf16(
                        false, aq[c], false, bb, (short)0, s, false, false);
            }
            sc[g] = s;
        }

        // ---- online softmax (C layout: VGPR i = row i + 8*half, lane = col)
#pragma unroll
        for (int i = 0; i < 8; ++i) {
            float mx = fmaxf(fmaxf(sc[0][i], sc[1][i]),
                             fmaxf(sc[2][i], sc[3][i]));
#pragma unroll
            for (int off = 8; off > 0; off >>= 1)
                mx = fmaxf(mx, __shfl_xor(mx, off, 32));   // 16-lane row max
            const float mnew  = fmaxf(m_run[i], mx);
            const float alpha = __expf(m_run[i] - mnew);
            m_run[i] = mnew;
            float ps = 0.f;
#pragma unroll
            for (int g = 0; g < 4; ++g) {
                const float p = __expf(sc[g][i] - mnew);
                ps += p;
                lds_p[wave][i + 8 * half][g * 16 + l16] = (__bf16)p;
            }
#pragma unroll
            for (int off = 8; off > 0; off >>= 1)
                ps += __shfl_xor(ps, off, 32);              // 16-lane row sum
            l_run[i] = alpha * l_run[i] + ps;
#pragma unroll
            for (int dg = 0; dg < 8; ++dg) o_acc[dg][i] *= alpha;
        }

        // ---- O += P V : reload P in A layout (same-wave LDS, in-order)
        v16bf ap[2];
#pragma unroll
        for (int c = 0; c < 2; ++c) {
            const __bf16* pp = &lds_p[wave][l16][32 * c + 8 * half];
            v8bf p0 = *(const v8bf*)pp;
            v8bf p1 = *(const v8bf*)(pp + 16);
#pragma unroll
            for (int j = 0; j < 8; ++j) { ap[c][j] = p0[j]; ap[c][8 + j] = p1[j]; }
        }
#pragma unroll
        for (int dg = 0; dg < 8; ++dg) {
            v8f o = o_acc[dg];
#pragma unroll
            for (int c = 0; c < 2; ++c) {
                const __bf16* vp = &lds_v[dg * 16 + l16][32 * c + 8 * half];
                v8bf b0 = *(const v8bf*)vp;
                v8bf b1 = *(const v8bf*)(vp + 16);
                v16bf bb;
#pragma unroll
                for (int j = 0; j < 8; ++j) { bb[j] = b0[j]; bb[8 + j] = b1[j]; }
                o = __builtin_amdgcn_wmma_f32_16x16x32_bf16(
                        false, ap[c], false, bb, (short)0, o, false, false);
            }
            o_acc[dg] = o;
        }

        // ---- retire async copies, publish tile t+1 into the bf16 buffers
        if (t + 1 < NTILES) {
            wait_asynccnt0();       // this wave's async writes to LDS landed
            __syncthreads();        // everyone done reading tile t + all writes
            convert_stage();
            __syncthreads();
        }
    }

    // ---- epilogue: normalize and store fp32 output [B,S,H,D]
#pragma unroll
    for (int i = 0; i < 8; ++i) {
        const float inv = 1.0f / l_run[i];
        const int row = qbase + wave * 16 + i + 8 * half;
        float* op = og + ((size_t)(b * S + row) * H + h) * D;
#pragma unroll
        for (int dg = 0; dg < 8; ++dg)
            op[dg * 16 + l16] = o_acc[dg][i] * inv;
    }
}

extern "C" void kernel_launch(void* const* d_in, const int* in_sizes, int n_in,
                              void* d_out, int out_size, void* d_ws, size_t ws_size,
                              hipStream_t stream) {
    (void)in_sizes; (void)n_in; (void)d_ws; (void)ws_size; (void)out_size;
    const float* q = (const float*)d_in[0];
    const float* k = (const float*)d_in[1];
    const float* v = (const float*)d_in[2];
    float* out = (float*)d_out;
    // allow >64KB dynamic LDS (gfx1250 WGP has 320KB)
    (void)hipFuncSetAttribute((const void*)ring_attn_fwd,
                              hipFuncAttributeMaxDynamicSharedMemorySize,
                              SMEM_BYTES);
    const int grid = B * H * (S / MT);   // 512 workgroups
    ring_attn_fwd<<<grid, 256, SMEM_BYTES, stream>>>(q, k, v, out);
}
